// BSquareModel_24326694765039
// MI455X (gfx1250) — matrix operand order repaired and checked by reading, not compiled, not run on hardware
//
#include <hip/hip_runtime.h>
#include <hip/hip_bf16.h>
#include <cstdint>
#include <cstddef>

#define NUM_CLASSES 10
#define BATCH       8192
#define IN_DIM      784
#define HID         128
#define NEXP        45
#define KP          800   // IN_DIM padded up to a multiple of 32
#define KTP         40    // LDS k-tile row length (elems): 80B rows -> 16B aligned, conflict-free
#define BM          64    // batch tile per workgroup (2 waves x 32 rows)
#define HPAD        136   // padded hidden row (272B = 68 dwords)

typedef __attribute__((ext_vector_type(16))) __bf16 v16bf;
typedef __attribute__((ext_vector_type(8)))  float  v8f;

union Frag { uint4 q[2]; v16bf v; };

// ---------------- prep kernels ----------------

__global__ __launch_bounds__(256) void zero_votes_k(float* __restrict__ v, int n) {
    int i = blockIdx.x * 256 + threadIdx.x;
    if (i < n) v[i] = 0.0f;
}

__global__ __launch_bounds__(256) void cvt_x_k(const float* __restrict__ x,
                                               __bf16* __restrict__ xb) {
    int i = blockIdx.x * 256 + threadIdx.x;      // over BATCH*KP
    if (i >= BATCH * KP) return;
    int b = i / KP, k = i % KP;
    float v = (k < IN_DIM) ? x[(size_t)b * IN_DIM + k] : 0.0f;
    xb[i] = (__bf16)v;
}

// W1[e][k][n] (fp32) -> w1t[e][n][k] (bf16, K padded to KP with zeros)
__global__ __launch_bounds__(256) void cvt_w1_k(const float* __restrict__ w1,
                                                __bf16* __restrict__ w1t) {
    int i = blockIdx.x * 256 + threadIdx.x;      // over NEXP*HID*KP
    if (i >= NEXP * HID * KP) return;
    int e = i / (HID * KP);
    int r = i % (HID * KP);
    int n = r / KP, k = r % KP;
    float v = (k < IN_DIM) ? w1[((size_t)e * IN_DIM + k) * HID + n] : 0.0f;
    w1t[i] = (__bf16)v;
}

// W2[e][k][n] (fp32) -> w2t[e][n][k] (bf16)
__global__ __launch_bounds__(256) void cvt_w2_k(const float* __restrict__ w2,
                                                __bf16* __restrict__ w2t) {
    int i = blockIdx.x * 256 + threadIdx.x;      // over NEXP*HID*HID
    if (i >= NEXP * HID * HID) return;
    int e = i / (HID * HID);
    int r = i % (HID * HID);
    int n = r / HID, k = r % HID;
    w2t[i] = (__bf16)w2[((size_t)e * HID + k) * HID + n];
}

// ---------------- async staging helpers (CDNA5 GLOBAL_LOAD_ASYNC_TO_LDS) ----------------
// Copies 64 contiguous bytes global->LDS per lane; INST_OFFSET applies to both addresses.
__device__ __forceinline__ void async_row64(const __bf16* g, unsigned ldsoff) {
    asm volatile(
        "global_load_async_to_lds_b128 %0, %1, off\n\t"
        "global_load_async_to_lds_b128 %0, %1, off offset:16\n\t"
        "global_load_async_to_lds_b128 %0, %1, off offset:32\n\t"
        "global_load_async_to_lds_b128 %0, %1, off offset:48"
        :: "v"(ldsoff), "v"(g) : "memory");
}

__device__ __forceinline__ void wait_async0() {
    asm volatile("s_wait_asynccnt 0x0" ::: "memory");
}

// ---------------- fused expert kernel ----------------
// block: 64 threads = 2 waves; each wave computes a 32x128 strip of the 64x128 tile,
// so each B fragment fetched from LDS feeds TWO v_wmma ops.
__global__ __launch_bounds__(64) void fused_mlp_vote_k(
    const __bf16* __restrict__ xb,   // [BATCH][KP] bf16
    const __bf16* __restrict__ w1t,  // [E][HID][KP] bf16 (N-major)
    const __bf16* __restrict__ w2t,  // [E][HID][HID] bf16 (N-major)
    const float*  __restrict__ b1,   // [E][HID]
    const float*  __restrict__ b2,   // [E][HID]
    const float*  __restrict__ Wout, // [E][HID][2]
    const float*  __restrict__ bout, // [E][2]
    float*        __restrict__ votes)// [BATCH][NUM_CLASSES]
{
    __shared__ __align__(16) __bf16 wtile[2][HID][KTP]; // double-buffered B k-tiles
    __shared__ __align__(16) __bf16 hs [BM][HPAD];      // layer-1 activations (bf16)
    __shared__ __align__(16) __bf16 hs2[BM][HPAD];      // layer-2 activations (bf16)
    __shared__ float b1s[HID], b2s[HID], wouts[2 * HID];

    const int e     = blockIdx.y;
    const int b0    = blockIdx.x * BM;
    const int tid   = threadIdx.x;       // 0..63
    const int lane  = tid & 31;
    const int wm    = (tid >> 5) * 32;   // wave's row base (0 or 32)
    const int lhalf = lane >> 4;         // 0: lanes 0-15, 1: lanes 16-31
    const int l16   = lane & 15;

    for (int c = tid; c < HID; c += 64) {
        b1s[c] = b1[e * HID + c];
        b2s[c] = b2[e * HID + c];
    }
    for (int c = tid; c < 2 * HID; c += 64)
        wouts[c] = Wout[e * 2 * HID + c];

    // LDS byte offsets for this thread's two staging rows (low 32 bits of generic ptr).
    const unsigned ldsA0 = (unsigned)(uintptr_t)&wtile[0][tid][0];
    const unsigned ldsB0 = (unsigned)(uintptr_t)&wtile[0][tid + 64][0];
    const unsigned ldsA1 = (unsigned)(uintptr_t)&wtile[1][tid][0];
    const unsigned ldsB1 = (unsigned)(uintptr_t)&wtile[1][tid + 64][0];

    v8f acc[16];
    #pragma unroll
    for (int t = 0; t < 16; ++t)
        #pragma unroll
        for (int j = 0; j < 8; ++j) acc[t][j] = 0.0f;

    // ---------------- layer 1: h1 = relu(x @ W1 + b1), K = 800 ----------------
    const __bf16* w1rowA = w1t + ((size_t)e * HID + tid) * KP;       // row N = tid
    const __bf16* w1rowB = w1rowA + (size_t)64 * KP;                 // row N = tid+64

    // A fragment bases: lanes 0-15 -> K 0-7 & 16-23; lanes 16-31 -> K 8-15 & 24-31
    const __bf16* xrow0 = xb + (size_t)(b0 + wm + l16) * KP + lhalf * 8;
    const __bf16* xrow1 = xrow0 + (size_t)16 * KP;

    const int NK1 = KP / 32; // 25
    async_row64(w1rowA, ldsA0);
    async_row64(w1rowB, ldsB0);

    for (int kt = 0; kt < NK1; ++kt) {
        const int buf = kt & 1;
        wait_async0();          // this wave's rows of tile kt are in LDS
        __syncthreads();        // everyone's rows are in LDS; all reads of buf^1 are done
        if (kt + 1 < NK1) {     // DMA tile kt+1 while we compute tile kt
            async_row64(w1rowA + (kt + 1) * 32, buf ? ldsA0 : ldsA1);
            async_row64(w1rowB + (kt + 1) * 32, buf ? ldsB0 : ldsB1);
        }
        Frag a0, a1;
        a0.q[0] = *(const uint4*)(xrow0 + kt * 32);
        a0.q[1] = *(const uint4*)(xrow0 + kt * 32 + 16);
        a1.q[0] = *(const uint4*)(xrow1 + kt * 32);
        a1.q[1] = *(const uint4*)(xrow1 + kt * 32 + 16);
        #pragma unroll
        for (int t = 0; t < 8; ++t) {
            Frag bf;
            const __bf16* bp = &wtile[buf][t * 16 + l16][lhalf * 8];
            bf.q[0] = ((const uint4*)bp)[0];
            bf.q[1] = *(const uint4*)(bp + 16);
            acc[t]     = __builtin_amdgcn_wmma_f32_16x16x32_bf16(
                false, a0.v, false, bf.v, (short)0, acc[t],     false, false);
            acc[8 + t] = __builtin_amdgcn_wmma_f32_16x16x32_bf16(
                false, a1.v, false, bf.v, (short)0, acc[8 + t], false, false);
        }
    }

    // epilogue 1: bias + relu -> bf16 (C/D layout: VGPR j, lanes<16 -> M=j, lanes>=16 -> M=8+j)
    #pragma unroll
    for (int g = 0; g < 2; ++g)
        #pragma unroll
        for (int t = 0; t < 8; ++t)
            #pragma unroll
            for (int j = 0; j < 8; ++j) {
                int r = wm + g * 16 + j + lhalf * 8;
                int c = t * 16 + l16;
                float v = acc[g * 8 + t][j] + b1s[c];
                hs[r][c] = (__bf16)fmaxf(v, 0.0f);
            }
    __syncthreads();            // hs published; all wtile reads of layer 1 done

    // ---------------- layer 2: h2 = relu(h1 @ W2 + b2), K = 128 ----------------
    #pragma unroll
    for (int t = 0; t < 16; ++t)
        #pragma unroll
        for (int j = 0; j < 8; ++j) acc[t][j] = 0.0f;

    const __bf16* w2rowA = w2t + ((size_t)e * HID + tid) * HID;
    const __bf16* w2rowB = w2rowA + (size_t)64 * HID;

    const int NK2 = HID / 32; // 4
    async_row64(w2rowA, ldsA0);
    async_row64(w2rowB, ldsB0);

    for (int kt = 0; kt < NK2; ++kt) {
        const int buf = kt & 1;
        wait_async0();
        __syncthreads();
        if (kt + 1 < NK2) {
            async_row64(w2rowA + (kt + 1) * 32, buf ? ldsA0 : ldsA1);
            async_row64(w2rowB + (kt + 1) * 32, buf ? ldsB0 : ldsB1);
        }
        Frag a0, a1;
        const __bf16* hp0 = &hs[wm + l16][kt * 32 + lhalf * 8];
        const __bf16* hp1 = &hs[wm + 16 + l16][kt * 32 + lhalf * 8];
        a0.q[0] = ((const uint4*)hp0)[0];
        a0.q[1] = *(const uint4*)(hp0 + 16);
        a1.q[0] = ((const uint4*)hp1)[0];
        a1.q[1] = *(const uint4*)(hp1 + 16);
        #pragma unroll
        for (int t = 0; t < 8; ++t) {
            Frag bf;
            const __bf16* bp = &wtile[buf][t * 16 + l16][lhalf * 8];
            bf.q[0] = ((const uint4*)bp)[0];
            bf.q[1] = *(const uint4*)(bp + 16);
            acc[t]     = __builtin_amdgcn_wmma_f32_16x16x32_bf16(
                false, a0.v, false, bf.v, (short)0, acc[t],     false, false);
            acc[8 + t] = __builtin_amdgcn_wmma_f32_16x16x32_bf16(
                false, a1.v, false, bf.v, (short)0, acc[8 + t], false, false);
        }
    }

    // epilogue 2
    #pragma unroll
    for (int g = 0; g < 2; ++g)
        #pragma unroll
        for (int t = 0; t < 8; ++t)
            #pragma unroll
            for (int j = 0; j < 8; ++j) {
                int r = wm + g * 16 + j + lhalf * 8;
                int c = t * 16 + l16;
                float v = acc[g * 8 + t][j] + b2s[c];
                hs2[r][c] = (__bf16)fmaxf(v, 0.0f);
            }
    __syncthreads();

    // ---------------- layer 3 (N=2) + vote ----------------
    {
        // triu(NUM_CLASSES, k=1) pair for expert e
        int i = 0, rem = e, span = NUM_CLASSES - 1;
        while (rem >= span) { rem -= span; --span; ++i; }
        const int c1 = i, c2 = i + 1 + rem;

        float o0 = bout[e * 2 + 0];
        float o1 = bout[e * 2 + 1];
        #pragma unroll 8
        for (int c = 0; c < HID; ++c) {
            float h = (float)hs2[tid][c];
            o0 += h * wouts[2 * c + 0];
            o1 += h * wouts[2 * c + 1];
        }
        const int cls = (o1 > o0) ? c2 : c1;   // argmax tie -> index 0 -> c1
        atomicAdd(votes + (size_t)(b0 + tid) * NUM_CLASSES + cls, 1.0f);
    }
}

// ---------------- launcher ----------------

extern "C" void kernel_launch(void* const* d_in, const int* in_sizes, int n_in,
                              void* d_out, int out_size, void* d_ws, size_t ws_size,
                              hipStream_t stream) {
    (void)in_sizes; (void)n_in; (void)out_size;
    const float* x    = (const float*)d_in[0];
    const float* W1   = (const float*)d_in[1];
    const float* b1   = (const float*)d_in[2];
    const float* W2   = (const float*)d_in[3];
    const float* b2   = (const float*)d_in[4];
    const float* Wout = (const float*)d_in[5];
    const float* bout = (const float*)d_in[6];
    float* votes = (float*)d_out;

    const size_t off_xb  = 0;
    const size_t off_w1t = off_xb  + (size_t)BATCH * KP  * sizeof(unsigned short);
    const size_t off_w2t = off_w1t + (size_t)NEXP * HID * KP  * sizeof(unsigned short);
    const size_t need    = off_w2t + (size_t)NEXP * HID * HID * sizeof(unsigned short);

    zero_votes_k<<<(BATCH * NUM_CLASSES + 255) / 256, 256, 0, stream>>>(
        votes, BATCH * NUM_CLASSES);

    if (ws_size < need) return;  // votes stay zeroed; no OOB writes

    __bf16* xb  = (__bf16*)((char*)d_ws + off_xb);
    __bf16* w1t = (__bf16*)((char*)d_ws + off_w1t);
    __bf16* w2t = (__bf16*)((char*)d_ws + off_w2t);

    cvt_x_k <<<(BATCH * KP + 255) / 256, 256, 0, stream>>>(x, xb);
    cvt_w1_k<<<(NEXP * HID * KP + 255) / 256, 256, 0, stream>>>(W1, w1t);
    cvt_w2_k<<<(NEXP * HID * HID + 255) / 256, 256, 0, stream>>>(W2, w2t);

    dim3 grid(BATCH / BM, NEXP);
    fused_mlp_vote_k<<<grid, 64, 0, stream>>>(xb, w1t, w2t, b1, b2, Wout, bout, votes);
}
